// MoE_MHA_89309549953084
// MI455X (gfx1250) — matrix operand order repaired and checked by reading, not compiled, not run on hardware
//
#include <hip/hip_runtime.h>

// ---------------------------------------------------------------------------
// MoE-GPT forward for MI455X (gfx1250, wave32, WMMA).
// GEMMs: v_wmma_f32_16x16x32_f16; tiles double-buffered in LDS via
// GLOBAL_LOAD_ASYNC_TO_LDS_B128 (ASYNCcnt) so DMA overlaps WMMA compute.
// Head-GEMM logits (524MB, write-once) use non-temporal stores to keep L2
// for the f16 weights.
// ---------------------------------------------------------------------------

typedef __attribute__((ext_vector_type(16))) _Float16 v16h;
typedef __attribute__((ext_vector_type(8)))  _Float16 v8h;
typedef __attribute__((ext_vector_type(4)))  _Float16 v4h;
typedef __attribute__((ext_vector_type(8)))  float    v8f;

#define C_DIM   768
#define T_DIM   1024
#define M_TOK   4096      // B*T
#define NEXP    8
#define VOCAB   32000
#define NLAYER  4
#define EPS_LN  1e-5f

// GEMM tiling: block = 256 threads = 8 waves; block tile 128(M) x 64(N), BK=32.
// Waves arranged 4(M) x 2(N); each wave owns a 32x32 tile = 2x2 WMMA frags.
#define BM 128
#define BN 64
#define BK 32
#define NKT (C_DIM / BK)  // 24 K-tiles (K fixed at 768 for every GEMM here)
#define LDS_PAD 40        // halves per row (32 data + 8 pad) -> 80B row stride

// ---------------------------------------------------------------------------
// helpers
// ---------------------------------------------------------------------------
__device__ __forceinline__ float wave_reduce_sum(float v) {
  #pragma unroll
  for (int off = 16; off > 0; off >>= 1) v += __shfl_xor(v, off, 32);
  return v;
}

// LDS byte offset of a __shared__ pointer (generic AS low 32 bits == LDS offset)
__device__ __forceinline__ unsigned int lds_off(const void* p) {
  return (unsigned int)(uintptr_t)p;
}

// Async memory->LDS 16B copy (GLOBAL_LOAD_ASYNC_TO_LDS_B128, ASYNCcnt)
__device__ __forceinline__ void async_copy_b128(unsigned int lds, const void* g) {
  asm volatile("global_load_async_to_lds_b128 %0, %1, off"
               :: "v"(lds), "v"((unsigned long long)(uintptr_t)g)
               : "memory");
}

// Async loads complete in order per wave: <=3 outstanding means the previous
// tile's 3 copies have landed while the next tile's 3 are still in flight.
__device__ __forceinline__ void wait_async_le3() {
  asm volatile("s_wait_asynccnt 3" ::: "memory");
}
__device__ __forceinline__ void wait_async0() {
  asm volatile("s_wait_asynccnt 0" ::: "memory");
}

// Stage one 128x32 f16 A tile and one 64x32 f16 B tile to LDS via async DMA.
// A: 128 rows x 4 chunks(16B) = 512 copies -> 2/thread.  B: 256 -> 1/thread.
// => 3 async instructions per wave per tile.
__device__ __forceinline__ void load_tiles_async(
    const _Float16* __restrict__ A, const _Float16* __restrict__ Bw,
    int bm, int bn, int k0, int tid,
    const _Float16* sA, const _Float16* sB) {
  #pragma unroll
  for (int it = 0; it < 2; ++it) {
    int v   = it * 256 + tid;
    int row = v >> 2;                  // 0..127
    int col = (v & 3) << 3;            // halves 0,8,16,24 (16B chunks)
    async_copy_b128(lds_off(&sA[row * LDS_PAD + col]),
                    A + (size_t)(bm + row) * C_DIM + k0 + col);
  }
  {
    int row = tid >> 2;                // 0..63
    int col = (tid & 3) << 3;
    async_copy_b128(lds_off(&sB[row * LDS_PAD + col]),
                    Bw + (size_t)(bn + row) * C_DIM + k0 + col);
  }
  if (k0 + 2 * BK < C_DIM) {           // HBM->L2 prefetch two tiles ahead
    __builtin_prefetch(A  + (size_t)(bm + (tid & 127)) * C_DIM + k0 + 2 * BK, 0, 0);
    __builtin_prefetch(Bw + (size_t)(bn + (tid &  63)) * C_DIM + k0 + 2 * BK, 0, 0);
  }
}

// One BK=32 step: 2 A-frags x 2 B-frags -> 4 WMMAs per wave.
// Lane layouts per CDNA5 ISA 7.12.2 (16-bit A 16x32, 16-bit B 32x16).
__device__ __forceinline__ void wmma_tile(const _Float16* sA, const _Float16* sB,
                                          int wm, int wn, int lrow, int lhi,
                                          v8f acc[2][2]) {
  v16h a[2], b[2];
  #pragma unroll
  for (int mf = 0; mf < 2; ++mf) {
    int ar = wm * 32 + mf * 16 + lrow;
    v8h lo = *(const v8h*)&sA[ar * LDS_PAD + lhi * 8];        // K 0..7 / 8..15
    v8h hi = *(const v8h*)&sA[ar * LDS_PAD + 16 + lhi * 8];   // K 16..23 / 24..31
    a[mf] = __builtin_shufflevector(lo, hi, 0,1,2,3,4,5,6,7,8,9,10,11,12,13,14,15);
  }
  #pragma unroll
  for (int nf = 0; nf < 2; ++nf) {
    int br = wn * 32 + nf * 16 + lrow;
    v8h lo = *(const v8h*)&sB[br * LDS_PAD + lhi * 8];        // K 0..7 / 8..15
    v8h hi = *(const v8h*)&sB[br * LDS_PAD + 16 + lhi * 8];   // K 16..23 / 24..31
    b[nf] = __builtin_shufflevector(lo, hi, 0,1,2,3,4,5,6,7,8,9,10,11,12,13,14,15);
  }
  #pragma unroll
  for (int mf = 0; mf < 2; ++mf)
    #pragma unroll
    for (int nf = 0; nf < 2; ++nf)
      acc[mf][nf] = __builtin_amdgcn_wmma_f32_16x16x32_f16(
          false, a[mf], false, b[nf], (short)0, acc[mf][nf], false, false);
}

// Double-buffered K pipeline shared by both GEMM kernels.
__device__ __forceinline__ void gemm_k_pipeline(
    const _Float16* __restrict__ A, const _Float16* __restrict__ Bw,
    int bm, int bn, int tid, int wm, int wn, int lrow, int lhi,
    _Float16 (*sA)[BM * LDS_PAD], _Float16 (*sB)[BN * LDS_PAD],
    v8f acc[2][2]) {
  load_tiles_async(A, Bw, bm, bn, 0, tid, sA[0], sB[0]);
  #pragma unroll 2
  for (int kt = 0; kt < NKT; ++kt) {
    const int cur = kt & 1;
    if (kt + 1 < NKT) {
      load_tiles_async(A, Bw, bm, bn, (kt + 1) * BK, tid, sA[cur ^ 1], sB[cur ^ 1]);
      wait_async_le3();                // tile kt landed; kt+1 still in flight
    } else {
      wait_async0();
    }
    __syncthreads();
    wmma_tile(sA[cur], sB[cur], wm, wn, lrow, lhi, acc);
    __syncthreads();                   // all waves done reading before reuse
  }
}

// ---------------------------------------------------------------------------
// generic GEMM: D[M,N] = A[M,768] (f16) x Bw[N,768]^T (f16) (+bias)(+residual)
// ---------------------------------------------------------------------------
__global__ __launch_bounds__(256) void gemm_f16_kernel(
    const _Float16* __restrict__ A, const _Float16* __restrict__ Bw,
    const float* __restrict__ bias, const float* residual, float* D,
    int N, int nt_store) {
  __shared__ _Float16 sA[2][BM * LDS_PAD];
  __shared__ _Float16 sB[2][BN * LDS_PAD];
  const int bm = blockIdx.y * BM, bn = blockIdx.x * BN;
  const int tid = threadIdx.x;
  const int wave = tid >> 5, lane = tid & 31;
  const int wm = wave & 3, wn = wave >> 2;
  const int lrow = lane & 15, lhi = lane >> 4;

  v8f acc[2][2] = {};
  gemm_k_pipeline(A, Bw, bm, bn, tid, wm, wn, lrow, lhi, sA, sB, acc);

  #pragma unroll
  for (int mf = 0; mf < 2; ++mf)
    #pragma unroll
    for (int nf = 0; nf < 2; ++nf) {
      const int colg = bn + wn * 32 + nf * 16 + lrow;
      const float bv = bias ? bias[colg] : 0.0f;
      #pragma unroll
      for (int i = 0; i < 8; ++i) {                 // VGPR i -> row mf*16+lhi*8+i
        const int rowg = bm + wm * 32 + mf * 16 + lhi * 8 + i;
        size_t off = (size_t)rowg * N + colg;
        float val = acc[mf][nf][i] + bv;
        if (residual) val += residual[off];
        if (nt_store) __builtin_nontemporal_store(val, &D[off]);  // logits: NT
        else          D[off] = val;
      }
    }
}

// ---------------------------------------------------------------------------
// MoE GEMM: H[m,c] += sum_e mask[m,e] * (A[m,:] . Ew[e][c,:] + Eb[e][c])
// ---------------------------------------------------------------------------
__global__ __launch_bounds__(256) void moe_gemm_kernel(
    const _Float16* __restrict__ A, const _Float16* __restrict__ Ew,
    const float* __restrict__ Eb, const float* __restrict__ mask,
    float* H, int N) {
  __shared__ _Float16 sA[2][BM * LDS_PAD];
  __shared__ _Float16 sB[2][BN * LDS_PAD];
  __shared__ float sMask[BM * NEXP];
  const int bm = blockIdx.y * BM, bn = blockIdx.x * BN;
  const int tid = threadIdx.x;
  const int wave = tid >> 5, lane = tid & 31;
  const int wm = wave & 3, wn = wave >> 2;
  const int lrow = lane & 15, lhi = lane >> 4;

  {                                           // stage 128x8 gate mask
    int r = tid >> 1, c4 = (tid & 1) << 2;
    float4 f = *(const float4*)(mask + (size_t)(bm + r) * NEXP + c4);
    *(float4*)&sMask[r * NEXP + c4] = f;
  }

  v8f yacc[2][2] = {};
  for (int e = 0; e < NEXP; ++e) {
    const _Float16* Bw = Ew + (size_t)e * C_DIM * C_DIM;
    v8f acc[2][2] = {};
    gemm_k_pipeline(A, Bw, bm, bn, tid, wm, wn, lrow, lhi, sA, sB, acc);
    #pragma unroll
    for (int mf = 0; mf < 2; ++mf)
      #pragma unroll
      for (int nf = 0; nf < 2; ++nf) {
        const int colg = bn + wn * 32 + nf * 16 + lrow;
        const float bv = Eb[(size_t)e * N + colg];
        #pragma unroll
        for (int i = 0; i < 8; ++i) {
          const int rowl = wm * 32 + mf * 16 + lhi * 8 + i;
          yacc[mf][nf][i] += sMask[rowl * NEXP + e] * (acc[mf][nf][i] + bv);
        }
      }
  }
  #pragma unroll
  for (int mf = 0; mf < 2; ++mf)
    #pragma unroll
    for (int nf = 0; nf < 2; ++nf) {
      const int colg = bn + wn * 32 + nf * 16 + lrow;
      #pragma unroll
      for (int i = 0; i < 8; ++i) {
        const int rowg = bm + wm * 32 + mf * 16 + lhi * 8 + i;
        size_t off = (size_t)rowg * N + colg;
        H[off] = H[off] + yacc[mf][nf][i];
      }
    }
}

// ---------------------------------------------------------------------------
// small kernels
// ---------------------------------------------------------------------------
__global__ void f32_to_f16_kernel(const float* __restrict__ s,
                                  _Float16* __restrict__ d, long n4) {
  long i = (long)blockIdx.x * 256 + threadIdx.x;
  long stride = (long)gridDim.x * 256;
  for (; i < n4; i += stride) {
    float4 f = *(const float4*)(s + i * 4);
    v4h h = {(_Float16)f.x, (_Float16)f.y, (_Float16)f.z, (_Float16)f.w};
    *(v4h*)(d + i * 4) = h;
  }
}

__global__ __launch_bounds__(256) void embed_kernel(
    const int* __restrict__ x, const float* __restrict__ tok,
    const float* __restrict__ pos, float* __restrict__ h) {
  const int row = blockIdx.x;
  const int tokid = x[row];
  const int t = row & (T_DIM - 1);
  for (int c = threadIdx.x; c < C_DIM; c += 256)
    h[(size_t)row * C_DIM + c] =
        tok[(size_t)tokid * C_DIM + c] + pos[(size_t)t * C_DIM + c];
}

// LayerNorm; writes fp32 (router input) and f16 (WMMA GEMM A operand) copies.
__global__ __launch_bounds__(256) void layernorm_kernel(
    const float* __restrict__ X, const float* __restrict__ g,
    const float* __restrict__ b, float* __restrict__ Yf,
    _Float16* __restrict__ Y16) {
  const int row = blockIdx.x, tid = threadIdx.x;
  const int wave = tid >> 5, lane = tid & 31;
  const float* x = X + (size_t)row * C_DIM;
  __shared__ float sm[8];
  __shared__ float mean_s, rstd_s;

  float v0 = x[tid], v1 = x[tid + 256], v2 = x[tid + 512];
  float s = wave_reduce_sum(v0 + v1 + v2);
  if (lane == 0) sm[wave] = s;
  __syncthreads();
  if (tid == 0) {
    float t = 0.f;
    for (int i = 0; i < 8; ++i) t += sm[i];
    mean_s = t * (1.0f / C_DIM);
  }
  __syncthreads();
  const float m = mean_s;
  float d0 = v0 - m, d1 = v1 - m, d2 = v2 - m;
  float q = wave_reduce_sum(d0 * d0 + d1 * d1 + d2 * d2);
  if (lane == 0) sm[wave] = q;
  __syncthreads();
  if (tid == 0) {
    float t = 0.f;
    for (int i = 0; i < 8; ++i) t += sm[i];
    rstd_s = rsqrtf(t * (1.0f / C_DIM) + EPS_LN);
  }
  __syncthreads();
  const float r = rstd_s;
  float y0 = d0 * r * g[tid]       + b[tid];
  float y1 = d1 * r * g[tid + 256] + b[tid + 256];
  float y2 = d2 * r * g[tid + 512] + b[tid + 512];
  float* yf = Yf + (size_t)row * C_DIM;
  yf[tid] = y0; yf[tid + 256] = y1; yf[tid + 512] = y2;
  _Float16* yh = Y16 + (size_t)row * C_DIM;
  yh[tid] = (_Float16)y0; yh[tid + 256] = (_Float16)y1; yh[tid + 512] = (_Float16)y2;
}

// Attention with batch_first=False semantics: L=4 sequence, N=1024 batch.
// One block per n; thread = (head, d); head = 64 threads = 2 waves.
// Output written as f16 (A operand of the out-proj WMMA GEMM).
__global__ __launch_bounds__(768) void attn_kernel(
    const float* __restrict__ qkv, _Float16* __restrict__ out) {
  const int n = blockIdx.x;
  const int t = threadIdx.x;
  const int head = t >> 6, d = t & 63;
  const int wave = t >> 5, lane = t & 31;

  float q[4], k[4], v[4];
  #pragma unroll
  for (int l = 0; l < 4; ++l) {
    size_t base = (size_t)(l * T_DIM + n) * (3 * C_DIM) + head * 64 + d;
    q[l] = qkv[base];
    k[l] = qkv[base + C_DIM];
    v[l] = qkv[base + 2 * C_DIM];
  }
  float s[16];
  #pragma unroll
  for (int l = 0; l < 4; ++l)
    #pragma unroll
    for (int m = 0; m < 4; ++m) s[l * 4 + m] = q[l] * k[m];
  #pragma unroll
  for (int j = 0; j < 16; ++j) s[j] = wave_reduce_sum(s[j]);

  __shared__ float red[24][16];
  if (lane == 0)
    #pragma unroll
    for (int j = 0; j < 16; ++j) red[wave][j] = s[j];
  __syncthreads();

  const int w0 = head << 1;
  #pragma unroll
  for (int l = 0; l < 4; ++l) {
    float sc[4], mx = -1e30f;
    #pragma unroll
    for (int m = 0; m < 4; ++m) {
      sc[m] = (red[w0][l * 4 + m] + red[w0 + 1][l * 4 + m]) * 0.125f; // Dh^-0.5
      mx = fmaxf(mx, sc[m]);
    }
    float sum = 0.f;
    #pragma unroll
    for (int m = 0; m < 4; ++m) { sc[m] = __expf(sc[m] - mx); sum += sc[m]; }
    const float inv = 1.0f / sum;
    float o = 0.f;
    #pragma unroll
    for (int m = 0; m < 4; ++m) o += sc[m] * inv * v[m];
    out[(size_t)(l * T_DIM + n) * C_DIM + head * 64 + d] = (_Float16)o;
  }
}

// gate mask: one wave per token row; mask[m,e] = (softmax(x.rw^T+rb)[e] > 0.5)
__global__ __launch_bounds__(256) void router_kernel(
    const float* __restrict__ X, const float* __restrict__ rw,
    const float* __restrict__ rb, float* __restrict__ mask) {
  const int row = blockIdx.x * 8 + (threadIdx.x >> 5);
  const int lane = threadIdx.x & 31;
  const float* x = X + (size_t)row * C_DIM;
  float lg[NEXP];
  #pragma unroll
  for (int e = 0; e < NEXP; ++e) {
    const float* w = rw + (size_t)e * C_DIM;
    float p = 0.f;
    for (int c = lane; c < C_DIM; c += 32) p += x[c] * w[c];
    lg[e] = wave_reduce_sum(p) + rb[e];
  }
  if (lane == 0) {
    float mx = lg[0];
    #pragma unroll
    for (int e = 1; e < NEXP; ++e) mx = fmaxf(mx, lg[e]);
    float sum = 0.f, ex[NEXP];
    #pragma unroll
    for (int e = 0; e < NEXP; ++e) { ex[e] = __expf(lg[e] - mx); sum += ex[e]; }
    const float inv = 1.0f / sum;
    #pragma unroll
    for (int e = 0; e < NEXP; ++e)
      mask[(size_t)row * NEXP + e] = (ex[e] * inv > 0.5f) ? 1.0f : 0.0f;
  }
}

// ---------------------------------------------------------------------------
// host driver
// ---------------------------------------------------------------------------
extern "C" void kernel_launch(void* const* d_in, const int* in_sizes, int n_in,
                              void* d_out, int out_size, void* d_ws, size_t ws_size,
                              hipStream_t stream) {
  const int*   x         = (const int*)d_in[0];
  const float* tok_emb   = (const float*)d_in[1];
  const float* pos_emb   = (const float*)d_in[2];
  const float* ln1_g     = (const float*)d_in[3];
  const float* ln1_b     = (const float*)d_in[4];
  const float* in_proj_w = (const float*)d_in[5];
  const float* in_proj_b = (const float*)d_in[6];
  const float* out_proj_w= (const float*)d_in[7];
  const float* out_proj_b= (const float*)d_in[8];
  const float* ln2_g     = (const float*)d_in[9];
  const float* ln2_b     = (const float*)d_in[10];
  const float* router_w  = (const float*)d_in[11];
  const float* router_b  = (const float*)d_in[12];
  const float* expert_w  = (const float*)d_in[13];
  const float* expert_b  = (const float*)d_in[14];
  const float* lnf_g     = (const float*)d_in[15];
  const float* lnf_b     = (const float*)d_in[16];
  const float* head_w    = (const float*)d_in[17];
  float* out = (float*)d_out;

  // workspace carve (256B aligned)
  uintptr_t p = (uintptr_t)d_ws;
  auto take = [&](size_t bytes) -> void* {
    uintptr_t r = p;
    p += (bytes + 255) & ~(size_t)255;
    return (void*)r;
  };
  const size_t n_ip = (size_t)NLAYER * 3 * C_DIM * C_DIM;
  const size_t n_op = (size_t)NLAYER * C_DIM * C_DIM;
  const size_t n_ex = (size_t)NLAYER * NEXP * C_DIM * C_DIM;
  const size_t n_hd = (size_t)VOCAB * C_DIM;
  _Float16* w_ip = (_Float16*)take(n_ip * 2);
  _Float16* w_op = (_Float16*)take(n_op * 2);
  _Float16* w_ex = (_Float16*)take(n_ex * 2);
  _Float16* w_hd = (_Float16*)take(n_hd * 2);
  float*    h    = (float*)take((size_t)M_TOK * C_DIM * 4);
  float*    hn   = (float*)take((size_t)M_TOK * C_DIM * 4);
  _Float16* hn16 = (_Float16*)take((size_t)M_TOK * C_DIM * 2);
  float*    qkv  = (float*)take((size_t)M_TOK * 3 * C_DIM * 4);
  _Float16* ao16 = (_Float16*)take((size_t)M_TOK * C_DIM * 2);
  float*    mask = (float*)take((size_t)M_TOK * NEXP * 4);

  // one-time (per launch) fp32 -> f16 weight conversion
  f32_to_f16_kernel<<<4096, 256, 0, stream>>>(in_proj_w,  w_ip, (long)(n_ip / 4));
  f32_to_f16_kernel<<<2048, 256, 0, stream>>>(out_proj_w, w_op, (long)(n_op / 4));
  f32_to_f16_kernel<<<8192, 256, 0, stream>>>(expert_w,   w_ex, (long)(n_ex / 4));
  f32_to_f16_kernel<<<8192, 256, 0, stream>>>(head_w,     w_hd, (long)(n_hd / 4));

  embed_kernel<<<M_TOK, 256, 0, stream>>>(x, tok_emb, pos_emb, h);

  const dim3 g_qkv (3 * C_DIM / BN, M_TOK / BM);   // (36, 32)
  const dim3 g_c   (C_DIM / BN,     M_TOK / BM);   // (12, 32)
  const dim3 g_head(VOCAB / BN,     M_TOK / BM);   // (500, 32)

  for (int L = 0; L < NLAYER; ++L) {
    layernorm_kernel<<<M_TOK, 256, 0, stream>>>(h, ln1_g + L * C_DIM,
                                                ln1_b + L * C_DIM, hn, hn16);
    gemm_f16_kernel<<<g_qkv, 256, 0, stream>>>(
        hn16, w_ip + (size_t)L * 3 * C_DIM * C_DIM, in_proj_b + L * 3 * C_DIM,
        nullptr, qkv, 3 * C_DIM, /*nt=*/0);
    attn_kernel<<<T_DIM, 768, 0, stream>>>(qkv, ao16);
    gemm_f16_kernel<<<g_c, 256, 0, stream>>>(
        ao16, w_op + (size_t)L * C_DIM * C_DIM, out_proj_b + L * C_DIM,
        /*residual=*/h, /*D=*/h, C_DIM, /*nt=*/0);
    layernorm_kernel<<<M_TOK, 256, 0, stream>>>(h, ln2_g + L * C_DIM,
                                                ln2_b + L * C_DIM, hn, hn16);
    router_kernel<<<M_TOK / 8, 256, 0, stream>>>(
        hn, router_w + (size_t)L * NEXP * C_DIM, router_b + L * NEXP, mask);
    moe_gemm_kernel<<<g_c, 256, 0, stream>>>(
        hn16, w_ex + (size_t)L * NEXP * C_DIM * C_DIM,
        expert_b + (size_t)L * NEXP * C_DIM, mask, h, C_DIM);
  }

  layernorm_kernel<<<M_TOK, 256, 0, stream>>>(h, lnf_g, lnf_b, hn, hn16);
  gemm_f16_kernel<<<g_head, 256, 0, stream>>>(hn16, w_hd, nullptr, nullptr, out,
                                              VOCAB, /*nt=*/1);
}
